// LoudnessEncoder_13589276524650
// MI455X (gfx1250) — matrix operand order repaired and checked by reading, not compiled, not run on hardware
//
#include <hip/hip_runtime.h>
#include <cmath>

typedef __attribute__((ext_vector_type(16))) _Float16 v16h;
typedef __attribute__((ext_vector_type(8)))  float    v8f;

#define T_SAMPLES 640000
#define HOP 128
#define NFFT 256
#define NBINS 129
#define NFRAMES 5001            // 1 + T/HOP
#define BATCH 64
#define FRAMES_PER_BLOCK 128    // 4 waves x 2 m-tiles x 16 frames
#define BLOCKS_PER_BATCH 40     // ceil(5001/128)
#define NT_TILES 9              // ceil(129/16)
#define KSTEPS 8                // 256/32
#define SPAN (FRAMES_PER_BLOCK*HOP + NFFT)     // 16640 floats = 66560 B dynamic LDS
#define BFRAG_HALFS (NT_TILES*KSTEPS*32*16)    // 36864 halfs per basis matrix

// ---------------------------------------------------------------------------
// Init: build windowed DFT basis directly in WMMA B-fragment layout.
// B fragment storage: [(nt*8 + kstep)*32 + lane] * 16 halfs (one v16h each).
// B-matrix 32x16 f16, contiguous k order used here (A matches): VGPR j, lanes 0-15 -> K=2j,2j+1 (col N=lane);
// lanes 16-31 -> K=16+2j,16+2j+1 (col N=lane-16).
// Window w[k] = 0.5 - 0.5*cos(2*pi*k/256) is folded in (diag(w) * B).
// Also computes mean of the A-weighting curve (a single scalar).
// ---------------------------------------------------------------------------
__global__ void loud_init(_Float16* __restrict__ Bc,
                          _Float16* __restrict__ Bs,
                          float* __restrict__ meanAw) {
    const int bid  = blockIdx.x;           // 0..71  (nt*8 + kstep)
    const int nt   = bid >> 3;
    const int s    = bid & 7;
    const int lane = threadIdx.x;          // 0..31
    const int n     = nt * 16 + (lane & 15);
    const int kbase = s * 32 + ((lane & 16) ? 16 : 0);
    const float step = 6.2831853071795864f / 256.0f;

    v16h vc, vs;
    #pragma unroll
    for (int j = 0; j < 8; ++j) {
        #pragma unroll
        for (int h = 0; h < 2; ++h) {
            int k = kbase + 2 * j + h;
            float w   = 0.5f - 0.5f * cosf((float)k * step);
            int   kn  = (k * n) & 255;              // exact arg reduction (period 256)
            float ang = (float)kn * step;
            vc[2 * j + h] = (_Float16)(w * cosf(ang));
            vs[2 * j + h] = (_Float16)(w * sinf(ang));
        }
    }
    size_t off = ((size_t)bid * 32 + lane) * 16;
    *(v16h*)(Bc + off) = vc;
    *(v16h*)(Bs + off) = vs;

    if (bid == 0 && lane == 0) {
        const double c0 = 12194.217 * 12194.217;
        const double c1 = 20.598997 * 20.598997;
        const double c2 = 107.65265 * 107.65265;
        const double c3 = 737.86223 * 737.86223;
        double acc = 0.0;
        for (int k = 0; k < NBINS; ++k) {
            double f  = (double)k * (16000.0 / 256.0);
            double f2 = f * f;
            double w  = 2.0 + 20.0 * (log10(c0) + 2.0 * log10(fmax(f2, 1e-30))
                        - log10(f2 + c0) - log10(f2 + c1)
                        - 0.5 * log10(f2 + c2) - 0.5 * log10(f2 + c3));
            if (w < -80.0) w = -80.0;
            acc += (double)(float)w;
        }
        meanAw[0] = (float)(acc / (double)NBINS);
    }
}

// ---------------------------------------------------------------------------
// Main: DFT-as-GEMM with v_wmma_f32_16x16x32_f16.
// Block = 128 threads (4 waves). Block stages 128 frames of signal into
// dynamic LDS via the CDNA5 async memory->LDS engine (ASYNCcnt-tracked,
// no VGPR bounce); each wave owns TWO 16-frame m-tiles so every loaded B
// fragment feeds 2 independent WMMAs (4 per cos/sin pair):
// 9 bin-tiles x 2 tiles x 2 (cos/sin) x 8 k-steps = 288 WMMAs per wave.
// ---------------------------------------------------------------------------
__global__ void __launch_bounds__(128) loud_main(const float* __restrict__ x,
                                                 const _Float16* __restrict__ Bc,
                                                 const _Float16* __restrict__ Bs,
                                                 const float* __restrict__ meanAwP,
                                                 float* __restrict__ out) {
    extern __shared__ float sx[];          // SPAN floats (66.5 KB, > 64KB static cap)

    const int b   = blockIdx.y;
    const int blk = blockIdx.x;
    const int tid = threadIdx.x;
    const float* xb = x + (size_t)b * T_SAMPLES;

    // Warm L2/L0 with the basis matrices (global_prefetch_b8); 1152 lines total.
    {
        const char* bbase = (const char*)Bc;            // Bc and Bs are contiguous
        const int nlines = (int)((2u * BFRAG_HALFS * sizeof(_Float16)) / 128u);
        for (int i = tid; i < nlines; i += 128)
            __builtin_prefetch(bbase + (size_t)i * 128, 0, 0);
    }

    // Stage signal tile with GLOBAL_LOAD_ASYNC_TO_LDS_B32 (reflect padding is a
    // per-lane address computation, which the async engine handles natively).
    // Flat LDS pointer: addr[31:0] == LDS byte offset (aperture rules).
    // SPAN = 130*128: uniform loop, EXEC stays all-ones.
    const int p0 = blk * (FRAMES_PER_BLOCK * HOP);     // padded-coordinate start
    for (int t = tid; t < SPAN; t += 128) {
        int j = p0 + t - (NFFT / 2);
        j = (j < 0) ? -j : j;
        if (j >= T_SAMPLES) j = 2 * (T_SAMPLES - 1) - j;
        unsigned int       loff  = (unsigned int)(uintptr_t)(const void*)(&sx[t]);
        unsigned long long gaddr = (unsigned long long)(uintptr_t)(const void*)(xb + j);
        asm volatile("global_load_async_to_lds_b32 %0, %1, off"
                     :: "v"(loff), "v"(gaddr)
                     : "memory");
    }
    asm volatile("s_wait_asynccnt 0x0" ::: "memory");   // this wave's async copies done
    __syncthreads();                                    // all waves' copies visible

    const int wave = tid >> 5;
    const int lane = tid & 31;
    const int m    = lane & 15;
    const int aoff = (lane & 16) ? 16 : 0;  // A layout: high lanes hold K-offset +16
    // Wave covers frames [blk*128 + wave*32, +32): tile0 rows 0..15, tile1 rows 16..31.
    const float* rowp0 = &sx[(wave * 32 + m) * HOP];
    const float* rowp1 = rowp0 + 16 * HOP;

    // Build A fragments (16x32 f16 per k-step) in the k order used for B here:
    // VGPR j in 0..7 -> K = 2j+aoff (contiguous, matches B).
    v16h a0[KSTEPS], a1[KSTEPS];
    #pragma unroll
    for (int s = 0; s < KSTEPS; ++s) {
        #pragma unroll
        for (int j = 0; j < 8; ++j) {
            int k = s * 32 + aoff + j * 2;
            float2 v0 = *(const float2*)(rowp0 + k);
            float2 v1 = *(const float2*)(rowp1 + k);
            a0[s][2 * j]     = (_Float16)v0.x;
            a0[s][2 * j + 1] = (_Float16)v0.y;
            a1[s][2 * j]     = (_Float16)v1.x;
            a1[s][2 * j + 1] = (_Float16)v1.y;
        }
    }

    float sums0[8] = {0.f, 0.f, 0.f, 0.f, 0.f, 0.f, 0.f, 0.f};
    float sums1[8] = {0.f, 0.f, 0.f, 0.f, 0.f, 0.f, 0.f, 0.f};

    for (int nt = 0; nt < NT_TILES; ++nt) {
        v8f accC0 = {0.f,0.f,0.f,0.f,0.f,0.f,0.f,0.f};
        v8f accS0 = {0.f,0.f,0.f,0.f,0.f,0.f,0.f,0.f};
        v8f accC1 = {0.f,0.f,0.f,0.f,0.f,0.f,0.f,0.f};
        v8f accS1 = {0.f,0.f,0.f,0.f,0.f,0.f,0.f,0.f};
        const v16h* bcf = (const v16h*)Bc + (size_t)(nt * KSTEPS) * 32 + lane;
        const v16h* bsf = (const v16h*)Bs + (size_t)(nt * KSTEPS) * 32 + lane;
        #pragma unroll
        for (int s = 0; s < KSTEPS; ++s) {
            v16h bc = bcf[s * 32];
            v16h bs = bsf[s * 32];
            // Each loaded fragment feeds two independent WMMA chains.
            accC0 = __builtin_amdgcn_wmma_f32_16x16x32_f16(
                        false, a0[s], false, bc, (short)0, accC0, false, false);
            accC1 = __builtin_amdgcn_wmma_f32_16x16x32_f16(
                        false, a1[s], false, bc, (short)0, accC1, false, false);
            accS0 = __builtin_amdgcn_wmma_f32_16x16x32_f16(
                        false, a0[s], false, bs, (short)0, accS0, false, false);
            accS1 = __builtin_amdgcn_wmma_f32_16x16x32_f16(
                        false, a1[s], false, bs, (short)0, accS1, false, false);
        }
        // C/D layout: VGPR r, lanes 0-15 -> M=r, N=lane; lanes 16-31 -> M=8+r.
        const int   n   = nt * 16 + m;
        const float msk = (n < NBINS) ? 1.0f : 0.0f;   // mask pad bins (select, no EXEC change)
        #pragma unroll
        for (int r = 0; r < 8; ++r) {
            float re0 = accC0[r], im0 = accS0[r];
            float re1 = accC1[r], im1 = accS1[r];
            sums0[r] += msk * __logf(sqrtf(re0 * re0 + im0 * im0) + 1e-7f);
            sums1[r] += msk * __logf(sqrtf(re1 * re1 + im1 * im1) + 1e-7f);
        }
    }

    // Reduce over the 16 lanes that share a frame row (xor within 16-lane halves).
    const float meanAw = meanAwP[0];
    #pragma unroll
    for (int r = 0; r < 8; ++r) {
        float v0 = sums0[r], v1 = sums1[r];
        v0 += __shfl_xor(v0, 1, 32);  v1 += __shfl_xor(v1, 1, 32);
        v0 += __shfl_xor(v0, 2, 32);  v1 += __shfl_xor(v1, 2, 32);
        v0 += __shfl_xor(v0, 4, 32);  v1 += __shfl_xor(v1, 4, 32);
        v0 += __shfl_xor(v0, 8, 32);  v1 += __shfl_xor(v1, 8, 32);
        sums0[r] = v0;  sums1[r] = v1;
    }
    if ((lane & 15) == 0) {
        const int mrow   = (lane >> 4) * 8;   // lane 0 -> rows 0..7, lane 16 -> rows 8..15
        const int frame0 = blk * FRAMES_PER_BLOCK + wave * 32 + mrow;
        #pragma unroll
        for (int r = 0; r < 8; ++r) {
            int f0 = frame0 + r;              // tile 0
            int f1 = frame0 + 16 + r;         // tile 1
            if (f0 < NFRAMES)
                out[(size_t)b * NFRAMES + f0] = sums0[r] * (1.0f / (float)NBINS) + meanAw;
            if (f1 < NFRAMES)
                out[(size_t)b * NFRAMES + f1] = sums1[r] * (1.0f / (float)NBINS) + meanAw;
        }
    }
}

// ---------------------------------------------------------------------------
extern "C" void kernel_launch(void* const* d_in, const int* in_sizes, int n_in,
                              void* d_out, int out_size, void* d_ws, size_t ws_size,
                              hipStream_t stream) {
    (void)in_sizes; (void)n_in; (void)out_size; (void)ws_size;
    const float* x = (const float*)d_in[0];
    float* out = (float*)d_out;

    _Float16* Bc     = (_Float16*)d_ws;
    _Float16* Bs     = Bc + BFRAG_HALFS;
    float*    meanAw = (float*)(Bs + BFRAG_HALFS);   // 32B-aligned region, ~144 KB total

    loud_init<<<NT_TILES * KSTEPS, 32, 0, stream>>>(Bc, Bs, meanAw);

    dim3 grid(BLOCKS_PER_BATCH, BATCH);
    loud_main<<<grid, 128, SPAN * sizeof(float), stream>>>(x, Bc, Bs, meanAw, out);
}
